// Transformer_MoE_FiLM_25975962206694
// MI455X (gfx1250) — matrix operand order repaired and checked by reading, not compiled, not run on hardware
//
#include <hip/hip_runtime.h>
#include <hip/hip_bf16.h>

// ---------------------------------------------------------------------------
// Types / helpers
// ---------------------------------------------------------------------------
typedef __attribute__((ext_vector_type(16))) __bf16 v16bf;
typedef __attribute__((ext_vector_type(8)))  float  v8f;

struct U32x8 { uint4 lo; uint4 hi; };

__device__ __forceinline__ unsigned short f2bf(float f) {
  unsigned int u = __builtin_bit_cast(unsigned int, f);
  u += 0x7FFFu + ((u >> 16) & 1u);               // round-to-nearest-even
  return (unsigned short)(u >> 16);
}
__device__ __forceinline__ float bf2f(unsigned short h) {
  unsigned int u = ((unsigned int)h) << 16;
  return __builtin_bit_cast(float, u);
}
__device__ __forceinline__ float gelu_f(float x) {
  return 0.5f * x * (1.0f + erff(x * 0.7071067811865475f));
}
__device__ __forceinline__ v16bf load_frag16(const unsigned short* lo,
                                             const unsigned short* hi) {
  U32x8 t;
  t.lo = *reinterpret_cast<const uint4*>(lo);
  t.hi = *reinterpret_cast<const uint4*>(hi);
  return __builtin_bit_cast(v16bf, t);
}
__device__ __forceinline__ v8f wmma_bf16(v16bf a, v16bf b, v8f c) {
  return __builtin_amdgcn_wmma_f32_16x16x32_bf16(false, a, false, b,
                                                 (short)0, c, false, false);
}
__device__ __forceinline__ v8f v8f_zero() {
  v8f z;
#pragma unroll
  for (int i = 0; i < 8; ++i) z[i] = 0.0f;
  return z;
}

// CDNA5 async global->LDS copy (ASYNCcnt-tracked). Generic LDS pointer's low
// 32 bits are the LDS byte offset (flat-aperture rule, ISA 10.2).
__device__ __forceinline__ unsigned lds_off(const void* p) {
  return (unsigned)(uintptr_t)p;
}
__device__ __forceinline__ void async_ld_b128(unsigned lds_addr, const void* g) {
  asm volatile("global_load_async_to_lds_b128 %0, %1, off"
               :: "v"(lds_addr), "v"((unsigned long long)(uintptr_t)g)
               : "memory");
}
__device__ __forceinline__ void wait_async0() {
  asm volatile("s_wait_asynccnt 0" ::: "memory");
}

// ---------------------------------------------------------------------------
// Generic bf16 GEMM with fused epilogues, double-buffered LDS, async staging.
//   C[M,N] = epi( A[M,K](bf16,row-major) @ B[K,N](bf16,row-major) + bias[N] )
// Block tile 128x128, staged K-step 64 (2 WMMA K-chunks), 256 threads
// (8 waves), wave tile 32x64.
// epi: 0 = f32 store            1 = bf16 store
//      2 = gelu -> bf16 store   3 = Cf (+)= rowscale[r]*(acc+bias)
//      4 = Cf += acc+bias  (residual, in-place f32)
// ---------------------------------------------------------------------------
#define GTM 128
#define GTN 128
#define GTK 64
#define GLK 80   // LDS row stride (elements): 160B = 16B-aligned + pad

__global__ __launch_bounds__(256) void k_gemm_bf16(
    const unsigned short* __restrict__ A, int lda,
    const unsigned short* __restrict__ Bw, int ldb,
    const float* __restrict__ bias,
    const float* __restrict__ rowscale, int rs_stride,
    float* __restrict__ Cf, unsigned short* __restrict__ Cb, int ldc,
    int K, int epi, int accum)
{
  __shared__ unsigned short As[2][GTM][GLK];
  __shared__ unsigned short Bt[2][GTN][GLK];   // B transposed: [n][k]

  const int tid  = threadIdx.x;
  const int lane = tid & 31;
  const int w    = tid >> 5;    // 0..7
  const int wm   = w >> 1;      // 0..3  (M waves)
  const int wn   = w & 1;       // 0..1  (N waves)
  const int m0   = blockIdx.y * GTM;
  const int n0   = blockIdx.x * GTN;

  v8f acc[2][4];
#pragma unroll
  for (int i = 0; i < 2; ++i)
#pragma unroll
    for (int j = 0; j < 4; ++j) acc[i][j] = v8f_zero();

  const int kbase = (lane >> 4) * 8;   // A fragment K base (lo/hi half)
  const int kb2   = (lane >> 4) * 16;  // unused; B fragments also use kbase

  // async-stage the A tile (128 x 64) for K offset k0 into buffer `buf`
  auto stageA = [&](int k0, int buf) {
#pragma unroll
    for (int i = 0; i < 4; ++i) {
      int v = i * 256 + tid;            // 0..1023 vec8 chunks
      int row = v >> 3, c8 = (v & 7) * 8;
      async_ld_b128(lds_off(&As[buf][row][c8]),
                    &A[(size_t)(m0 + row) * lda + k0 + c8]);
    }
  };
  // load B tile (64 x 128) into registers
  auto loadB = [&](int k0, uint4* r4) {
#pragma unroll
    for (int i = 0; i < 4; ++i) {
      int v = i * 256 + tid;
      int kr = v >> 4, nc = (v & 15) * 8;
      r4[i] = *reinterpret_cast<const uint4*>(
          &Bw[(size_t)(k0 + kr) * ldb + n0 + nc]);
    }
  };
  // scatter register B tile transposed into Bt[buf]
  auto storeB = [&](int buf, const uint4* r4) {
#pragma unroll
    for (int i = 0; i < 4; ++i) {
      int v = i * 256 + tid;
      int kr = v >> 4, nc = (v & 15) * 8;
      unsigned short e[8];
      *reinterpret_cast<uint4*>(e) = r4[i];
#pragma unroll
      for (int j = 0; j < 8; ++j) Bt[buf][nc + j][kr] = e[j];
    }
  };
  // 16 WMMAs over one staged 64-deep tile; B fragments batched so the
  // scheduler can retire WMMAs against partial DScnt instead of 0-waits
  auto compute = [&](int buf) {
#pragma unroll
    for (int kc = 0; kc < 2; ++kc) {
      const int ko = kc * 32;
      v16bf afrag[2];
#pragma unroll
      for (int i = 0; i < 2; ++i) {
        int r = wm * 32 + i * 16 + (lane & 15);
        afrag[i] = load_frag16(&As[buf][r][ko + kbase],
                               &As[buf][r][ko + kbase + 16]);
      }
      v16bf bfrag[4];
#pragma unroll
      for (int j = 0; j < 4; ++j) {
        int c = wn * 64 + j * 16 + (lane & 15);
        bfrag[j] = load_frag16(&Bt[buf][c][ko + kbase],
                               &Bt[buf][c][ko + kbase + 16]);
      }
#pragma unroll
      for (int j = 0; j < 4; ++j)
#pragma unroll
        for (int i = 0; i < 2; ++i)
          acc[i][j] = wmma_bf16(afrag[i], bfrag[j], acc[i][j]);
    }
  };

  // prologue: stage tile 0
  {
    uint4 r4[4];
    stageA(0, 0);
    loadB(0, r4);
    storeB(0, r4);
    wait_async0();
  }
  __syncthreads();

  int buf = 0;
  for (int k0 = 0; k0 < K; k0 += GTK) {
    const int nb = buf ^ 1;
    const bool more = (k0 + GTK) < K;
    uint4 r4[4];
    if (more) {
      stageA(k0 + GTK, nb);   // async copy overlaps compute below
      loadB(k0 + GTK, r4);    // global loads overlap compute below
    }
    compute(buf);
    if (more) {
      storeB(nb, r4);
      wait_async0();
      __syncthreads();
    }
    buf = nb;
  }

  // ---- epilogue ----
#pragma unroll
  for (int i = 0; i < 2; ++i) {
#pragma unroll
    for (int j = 0; j < 4; ++j) {
#pragma unroll
      for (int v = 0; v < 8; ++v) {
        int r = m0 + wm * 32 + i * 16 + 8 * (lane >> 4) + v;
        int c = n0 + wn * 64 + j * 16 + (lane & 15);
        float x = acc[i][j][v] + bias[c];
        size_t idx = (size_t)r * ldc + c;
        if (epi == 0) {
          Cf[idx] = x;
        } else if (epi == 1) {
          Cb[idx] = f2bf(x);
        } else if (epi == 2) {
          Cb[idx] = f2bf(gelu_f(x));
        } else if (epi == 3) {
          float s = rowscale[(size_t)r * rs_stride];
          Cf[idx] = (accum ? Cf[idx] : 0.0f) + s * x;
        } else {  // 4: residual add in place
          Cf[idx] = Cf[idx] + x;
        }
      }
    }
  }
}

// ---------------------------------------------------------------------------
// Flash attention: one block per (b*H+h, q-tile of 128). 8 waves x 16 q-rows.
// qkv layout: [token][3072] bf16, q at h*64, k at 1024+h*64, v at 2048+h*64.
// Output o: [token][1024] bf16.
// ---------------------------------------------------------------------------
__global__ __launch_bounds__(256) void k_attn(
    const unsigned short* __restrict__ qkv, unsigned short* __restrict__ o)
{
  __shared__ unsigned short Qs[128][72];
  __shared__ unsigned short Ks[64][72];       // K row-major == B^T layout
  __shared__ unsigned short Vt[64][72];       // V transposed [d][key]
  __shared__ unsigned short Ps[8][16][72];    // per-wave P tile

  const int tid  = threadIdx.x;
  const int lane = tid & 31;
  const int w    = tid >> 5;
  const int bh   = blockIdx.x;           // 0..63
  const int b    = bh >> 4, h = bh & 15;
  const int tok0 = b * 1024 + blockIdx.y * 128;
  const int qcol = h * 64;
  const int kcol = 1024 + h * 64;
  const int vcol = 2048 + h * 64;
  const float scale = 0.125f;            // DH^-0.5, DH=64

  // async-stage Q tile (128x64)
#pragma unroll
  for (int i = 0; i < 4; ++i) {
    int v = i * 256 + tid;
    int row = v >> 3, c8 = (v & 7) * 8;
    async_ld_b128(lds_off(&Qs[row][c8]),
                  &qkv[(size_t)(tok0 + row) * 3072 + qcol + c8]);
  }
  wait_async0();
  __syncthreads();

  const int kbase = (lane >> 4) * 8;
  const int kb2   = (lane >> 4) * 16;
  const int qr0   = w * 16;

  v16bf qfrag[2];
#pragma unroll
  for (int i = 0; i < 2; ++i) {
    int r = qr0 + (lane & 15);
    qfrag[i] = load_frag16(&Qs[r][i * 32 + kbase], &Qs[r][i * 32 + kbase + 16]);
  }

  float mrow[8], lrow[8];
#pragma unroll
  for (int v = 0; v < 8; ++v) { mrow[v] = -1e30f; lrow[v] = 0.0f; }
  v8f oacc[4];
#pragma unroll
  for (int j = 0; j < 4; ++j) oacc[j] = v8f_zero();

  for (int kb = 0; kb < 1024; kb += 64) {
    __syncthreads();   // previous iteration's reads of Ks/Vt finished
    // async-stage K block (64x64) row-major
#pragma unroll
    for (int i = 0; i < 2; ++i) {
      int v = i * 256 + tid;
      int row = v >> 3, c8 = (v & 7) * 8;
      async_ld_b128(lds_off(&Ks[row][c8]),
                    &qkv[(size_t)(b * 1024 + kb + row) * 3072 + kcol + c8]);
    }
    // stage V block transposed: Vt[d][key] (manual: transpose on store)
#pragma unroll
    for (int i = 0; i < 2; ++i) {
      int v = i * 256 + tid;
      int row = v >> 3, c8 = (v & 7) * 8;   // row = key, c8 = d
      uint4 d4 = *reinterpret_cast<const uint4*>(
          &qkv[(size_t)(b * 1024 + kb + row) * 3072 + vcol + c8]);
      unsigned short e[8];
      *reinterpret_cast<uint4*>(e) = d4;
#pragma unroll
      for (int j = 0; j < 8; ++j) Vt[c8 + j][row] = e[j];
    }
    wait_async0();
    __syncthreads();

    // S = Q @ K^T  (16 q-rows x 64 keys)
    v8f sacc[4];
#pragma unroll
    for (int j = 0; j < 4; ++j) sacc[j] = v8f_zero();
#pragma unroll
    for (int j = 0; j < 4; ++j) {
      int c = j * 16 + (lane & 15);
#pragma unroll
      for (int i = 0; i < 2; ++i) {
        v16bf kf = load_frag16(&Ks[c][i * 32 + kbase], &Ks[c][i * 32 + kbase + 16]);
        sacc[j] = wmma_bf16(qfrag[i], kf, sacc[j]);
      }
    }

    // online softmax; row stats reduced over the 16-lane half
    float alpha[8];
#pragma unroll
    for (int v = 0; v < 8; ++v) {
      float mx = sacc[0][v];
#pragma unroll
      for (int j = 1; j < 4; ++j) mx = fmaxf(mx, sacc[j][v]);
      for (int off = 1; off < 16; off <<= 1)
        mx = fmaxf(mx, __shfl_xor(mx, off, 32));
      float mn = fmaxf(mrow[v], mx * scale);
      alpha[v] = __expf(mrow[v] - mn);
      mrow[v] = mn;
    }
#pragma unroll
    for (int v = 0; v < 8; ++v) {
      float rs = 0.0f;
      int pr = 8 * (lane >> 4) + v;
#pragma unroll
      for (int j = 0; j < 4; ++j) {
        float p = __expf(sacc[j][v] * scale - mrow[v]);
        rs += p;
        Ps[w][pr][j * 16 + (lane & 15)] = f2bf(p);
      }
      for (int off = 1; off < 16; off <<= 1) rs += __shfl_xor(rs, off, 32);
      lrow[v] = lrow[v] * alpha[v] + rs;
    }
#pragma unroll
    for (int j = 0; j < 4; ++j)
#pragma unroll
      for (int v = 0; v < 8; ++v) oacc[j][v] *= alpha[v];

    // O += P @ V   (P: 16x64 in Ps, V^T in Vt)
#pragma unroll
    for (int i = 0; i < 2; ++i) {
      int r = (lane & 15);
      v16bf pf = load_frag16(&Ps[w][r][i * 32 + kbase],
                             &Ps[w][r][i * 32 + kbase + 16]);
#pragma unroll
      for (int j = 0; j < 4; ++j) {
        int c = j * 16 + (lane & 15);
        v16bf vf = load_frag16(&Vt[c][i * 32 + kbase], &Vt[c][i * 32 + kbase + 16]);
        oacc[j] = wmma_bf16(pf, vf, oacc[j]);
      }
    }
  }

  // write O / l
#pragma unroll
  for (int j = 0; j < 4; ++j)
#pragma unroll
    for (int v = 0; v < 8; ++v) {
      int row = qr0 + 8 * (lane >> 4) + v;
      int c = j * 16 + (lane & 15);
      o[(size_t)(tok0 + row) * 1024 + h * 64 + c] = f2bf(oacc[j][v] / lrow[v]);
    }
}

// ---------------------------------------------------------------------------
// LayerNorm over D=1024, one block per token, bf16 output
// ---------------------------------------------------------------------------
__global__ __launch_bounds__(256) void k_ln(
    const float* __restrict__ X, const float* __restrict__ g,
    const float* __restrict__ bt, unsigned short* __restrict__ out)
{
  __shared__ float rs[8], rs2[8];
  const int tok = blockIdx.x;
  const float* xr = X + (size_t)tok * 1024;
  float vals[4], s = 0.0f, s2 = 0.0f;
#pragma unroll
  for (int i = 0; i < 4; ++i) {
    int d = threadIdx.x + i * 256;
    float v = xr[d];
    vals[i] = v; s += v; s2 += v * v;
  }
  for (int off = 1; off < 32; off <<= 1) {
    s += __shfl_xor(s, off, 32);
    s2 += __shfl_xor(s2, off, 32);
  }
  int w = threadIdx.x >> 5, lane = threadIdx.x & 31;
  if (lane == 0) { rs[w] = s; rs2[w] = s2; }
  __syncthreads();
  if (w == 0) {
    float a = (lane < 8) ? rs[lane] : 0.0f;
    float a2 = (lane < 8) ? rs2[lane] : 0.0f;
    for (int off = 1; off < 8; off <<= 1) {
      a += __shfl_xor(a, off, 32);
      a2 += __shfl_xor(a2, off, 32);
    }
    if (lane == 0) { rs[0] = a; rs2[0] = a2; }
  }
  __syncthreads();
  float mean = rs[0] * (1.0f / 1024.0f);
  float var = rs2[0] * (1.0f / 1024.0f) - mean * mean;
  float inv = rsqrtf(var + 1e-5f);
#pragma unroll
  for (int i = 0; i < 4; ++i) {
    int d = threadIdx.x + i * 256;
    out[(size_t)tok * 1024 + d] = f2bf((vals[i] - mean) * inv * g[d] + bt[d]);
  }
}

// ---------------------------------------------------------------------------
// Router: logits = X @ Wr + br  (one wave per token)
// ---------------------------------------------------------------------------
__global__ __launch_bounds__(256) void k_router(
    const float* __restrict__ X, const float* __restrict__ Wr,
    const float* __restrict__ br, float* __restrict__ logits)
{
  int w = threadIdx.x >> 5, lane = threadIdx.x & 31;
  int tok = blockIdx.x * 8 + w;
  const float* xr = X + (size_t)tok * 1024;
  float acc[8];
#pragma unroll
  for (int e = 0; e < 8; ++e) acc[e] = 0.0f;
  for (int d = lane; d < 1024; d += 32) {
    float xv = xr[d];
    const float* wr = Wr + (size_t)d * 8;
#pragma unroll
    for (int e = 0; e < 8; ++e) acc[e] += xv * wr[e];
  }
#pragma unroll
  for (int e = 0; e < 8; ++e)
    for (int off = 1; off < 32; off <<= 1) acc[e] += __shfl_xor(acc[e], off, 32);
  if (lane == 0) {
#pragma unroll
    for (int e = 0; e < 8; ++e) logits[(size_t)tok * 8 + e] = acc[e] + br[e];
  }
}

// top-2 keep + softmax; writes ws copy and output copy
__global__ __launch_bounds__(256) void k_top2(
    const float* __restrict__ logits, float* __restrict__ wts,
    float* __restrict__ out_w)
{
  int tok = blockIdx.x * 256 + threadIdx.x;
  if (tok >= 4096) return;
  float lg[8], m1 = -1e30f, m2 = -1e30f;
#pragma unroll
  for (int e = 0; e < 8; ++e) {
    float v = logits[(size_t)tok * 8 + e];
    lg[e] = v;
    if (v > m1) { m2 = m1; m1 = v; }
    else if (v > m2) m2 = v;
  }
  float den = 0.0f, ex[8];
#pragma unroll
  for (int e = 0; e < 8; ++e) {
    float v = (lg[e] < m2) ? 0.0f : __expf(lg[e] - m1);
    ex[e] = v; den += v;
  }
  float inv = 1.0f / den;
#pragma unroll
  for (int e = 0; e < 8; ++e) {
    float wv = ex[e] * inv;
    wts[(size_t)tok * 8 + e] = wv;
    out_w[(size_t)tok * 8 + e] = wv;
  }
}

// ---------------------------------------------------------------------------
// Elementwise kernels
// ---------------------------------------------------------------------------
__global__ __launch_bounds__(256) void k_f2bf_n(
    const float* __restrict__ s, unsigned short* __restrict__ d, int n)
{
#pragma unroll
  for (int j = 0; j < 4; ++j) {
    int i = blockIdx.x * 1024 + j * 256 + threadIdx.x;
    if (i < n) d[i] = f2bf(s[i]);
  }
}

__global__ __launch_bounds__(256) void k_copy_f32(
    const float* __restrict__ s, float* __restrict__ d, int n)
{
  int i = blockIdx.x * 256 + threadIdx.x;
  if (i < n) d[i] = s[i];
}

// ytf = gelu(U) * V  -> bf16
__global__ __launch_bounds__(256) void k_glu(
    const float* __restrict__ U, const float* __restrict__ V,
    unsigned short* __restrict__ ytf, int n)
{
#pragma unroll
  for (int j = 0; j < 4; ++j) {
    int i = blockIdx.x * 1024 + j * 256 + threadIdx.x;
    if (i < n) ytf[i] = f2bf(gelu_f(U[i]) * V[i]);
  }
}

// y = w0*gelu(ytf) + Pacc*ytf + Aacc  -> bf16
__global__ __launch_bounds__(256) void k_combine(
    const unsigned short* __restrict__ ytf, const float* __restrict__ P,
    const float* __restrict__ A, const float* __restrict__ wts,
    unsigned short* __restrict__ y, int n)
{
#pragma unroll
  for (int j = 0; j < 4; ++j) {
    int i = blockIdx.x * 1024 + j * 256 + threadIdx.x;
    if (i < n) {
      float t = bf2f(ytf[i]);
      int tok = i >> 10;
      float w0 = wts[(size_t)tok * 8];
      y[i] = f2bf(w0 * gelu_f(t) + P[i] * t + A[i]);
    }
  }
}

// ---------------------------------------------------------------------------
// Host orchestration
// ---------------------------------------------------------------------------
extern "C" void kernel_launch(void* const* d_in, const int* in_sizes, int n_in,
                              void* d_out, int out_size, void* d_ws,
                              size_t ws_size, hipStream_t stream)
{
  (void)in_sizes; (void)n_in; (void)out_size; (void)ws_size;
  constexpr int BN = 4096, D = 1024, I3 = 3072, MLPd = 4096, E = 8, L = 2;

  const float* in_x  = (const float*)d_in[0];
  const float* ln1_g = (const float*)d_in[1];
  const float* ln1_b = (const float*)d_in[2];
  const float* Wqkv  = (const float*)d_in[3];
  const float* bqkv  = (const float*)d_in[4];
  const float* Wo    = (const float*)d_in[5];
  const float* bo    = (const float*)d_in[6];
  const float* Wr    = (const float*)d_in[7];
  const float* br    = (const float*)d_in[8];
  const float* Wa1   = (const float*)d_in[9];
  const float* ba1   = (const float*)d_in[10];
  const float* Wa2   = (const float*)d_in[11];
  const float* ba2   = (const float*)d_in[12];
  const float* ln2_g = (const float*)d_in[13];
  const float* ln2_b = (const float*)d_in[14];
  const float* Wb0   = (const float*)d_in[15];
  const float* bb0   = (const float*)d_in[16];
  const float* Wb1   = (const float*)d_in[17];
  const float* bb1   = (const float*)d_in[18];
  const float* We_p  = (const float*)d_in[19];
  const float* be_p  = (const float*)d_in[20];
  const float* We_a  = (const float*)d_in[21];
  const float* be_a  = (const float*)d_in[22];

  char* p = (char*)d_ws;
  auto alloc = [&](size_t bytes) -> void* {
    void* r = (void*)p;
    p += (bytes + 255) & ~(size_t)255;
    return r;
  };
  float*          X     = (float*)alloc((size_t)BN * D * 4);
  unsigned short* hb    = (unsigned short*)alloc((size_t)BN * D * 2);
  unsigned short* qkvb  = (unsigned short*)alloc((size_t)BN * I3 * 2);
  unsigned short* ob    = (unsigned short*)alloc((size_t)BN * D * 2);
  float*          U     = (float*)alloc((size_t)BN * D * 4);  // also Pacc
  float*          V     = (float*)alloc((size_t)BN * D * 4);  // also Aacc
  unsigned short* ytfb  = (unsigned short*)alloc((size_t)BN * D * 2);
  unsigned short* yb    = (unsigned short*)alloc((size_t)BN * D * 2);
  unsigned short* a1b   = (unsigned short*)alloc((size_t)BN * MLPd * 2);
  float*          lgts  = (float*)alloc((size_t)BN * E * 4);
  float*          wts   = (float*)alloc((size_t)BN * E * 4);
  unsigned short* wbqkv = (unsigned short*)alloc((size_t)D * I3 * 2);
  unsigned short* wbo   = (unsigned short*)alloc((size_t)D * D * 2);
  unsigned short* wb0   = (unsigned short*)alloc((size_t)D * D * 2);
  unsigned short* wb1   = (unsigned short*)alloc((size_t)D * D * 2);
  unsigned short* wbp   = (unsigned short*)alloc((size_t)7 * D * D * 2);
  unsigned short* wba   = (unsigned short*)alloc((size_t)7 * D * D * 2);
  unsigned short* wba1  = (unsigned short*)alloc((size_t)D * MLPd * 2);
  unsigned short* wba2  = (unsigned short*)alloc((size_t)MLPd * D * 2);

  auto conv = [&](const float* s, unsigned short* d, size_t n) {
    k_f2bf_n<<<(unsigned)((n + 1023) / 1024), 256, 0, stream>>>(s, d, (int)n);
  };
  auto gemm = [&](const unsigned short* A, int lda, const unsigned short* Bw,
                  int ldb, const float* bias, const float* rsc, int rss,
                  float* Cf, unsigned short* Cb, int ldc, int M, int N, int K,
                  int epi, int accum) {
    dim3 g(N / 128, M / 128);
    k_gemm_bf16<<<g, 256, 0, stream>>>(A, lda, Bw, ldb, bias, rsc, rss, Cf,
                                       Cb, ldc, K, epi, accum);
  };

  // X <- input
  k_copy_f32<<<BN * D / 256, 256, 0, stream>>>(in_x, X, BN * D);

  for (int l = 0; l < L; ++l) {
    // per-layer weight conversion (region reused across layers)
    conv(Wqkv + (size_t)l * D * I3, wbqkv, (size_t)D * I3);
    conv(Wo + (size_t)l * D * D, wbo, (size_t)D * D);
    conv(Wb0 + (size_t)l * D * D, wb0, (size_t)D * D);
    conv(Wb1 + (size_t)l * D * D, wb1, (size_t)D * D);
    conv(We_p + (size_t)l * 7 * D * D, wbp, (size_t)7 * D * D);
    conv(We_a + (size_t)l * 7 * D * D, wba, (size_t)7 * D * D);
    conv(Wa1 + (size_t)l * D * MLPd, wba1, (size_t)D * MLPd);
    conv(Wa2 + (size_t)l * MLPd * D, wba2, (size_t)MLPd * D);

    // attention block
    k_ln<<<BN, 256, 0, stream>>>(X, ln1_g + l * D, ln1_b + l * D, hb);
    gemm(hb, D, wbqkv, I3, bqkv + (size_t)l * I3, nullptr, 0, nullptr, qkvb,
         I3, BN, I3, D, /*epi=*/1, 0);
    k_attn<<<dim3(64, 8), 256, 0, stream>>>(qkvb, ob);
    gemm(ob, D, wbo, D, bo + (size_t)l * D, nullptr, 0, X, nullptr, D, BN, D,
         D, /*epi=*/4, 0);   // X += o@Wo + bo

    // router
    k_router<<<BN / 8, 256, 0, stream>>>(X, Wr + (size_t)l * D * E,
                                         br + (size_t)l * E, lgts);
    k_top2<<<BN / 256, 256, 0, stream>>>(
        lgts, wts, (float*)d_out + (size_t)BN * D + (size_t)l * BN * E);

    // FiLM trunk: ytf = gelu(yt@W0+b0) * (yt@W1+b1)
    k_ln<<<BN, 256, 0, stream>>>(X, ln2_g + l * D, ln2_b + l * D, hb);
    gemm(hb, D, wb0, D, bb0 + (size_t)l * D, nullptr, 0, U, nullptr, D, BN, D,
         D, /*epi=*/0, 0);
    gemm(hb, D, wb1, D, bb1 + (size_t)l * D, nullptr, 0, V, nullptr, D, BN, D,
         D, /*epi=*/0, 0);
    k_glu<<<BN * D / 1024, 256, 0, stream>>>(U, V, ytfb, BN * D);

    // experts: Pacc(=U) = sum_e w[e+1]*(ytf@Pw_e + Pb_e); Aacc(=V) likewise
    for (int e = 0; e < 7; ++e) {
      gemm(ytfb, D, wbp + (size_t)e * D * D,
           D, be_p + ((size_t)l * 7 + e) * D, wts + (e + 1), 8, U, nullptr, D,
           BN, D, D, /*epi=*/3, e > 0);
      gemm(ytfb, D, wba + (size_t)e * D * D,
           D, be_a + ((size_t)l * 7 + e) * D, wts + (e + 1), 8, V, nullptr, D,
           BN, D, D, /*epi=*/3, e > 0);
    }
    k_combine<<<BN * D / 1024, 256, 0, stream>>>(ytfb, U, V, wts, yb, BN * D);

    // adaptor MLP + residual
    gemm(yb, D, wba1, MLPd, ba1 + (size_t)l * MLPd, nullptr, 0, nullptr, a1b,
         MLPd, BN, MLPd, D, /*epi=*/2, 0);
    gemm(a1b, MLPd, wba2, D, ba2 + (size_t)l * D, nullptr, 0, X, nullptr, D,
         BN, D, MLPd, /*epi=*/4, 0);   // X += gelu(...)@Wa2 + ba2
  }

  // final x -> d_out
  k_copy_f32<<<BN * D / 256, 256, 0, stream>>>(X, (float*)d_out, BN * D);
}
